// TransMIL_68582037782743
// MI455X (gfx1250) — compile-verified, matrix operand order checked by reading, not run on hardware
//
#include <hip/hip_runtime.h>
#include <hip/hip_bf16.h>
#include <math.h>

typedef __attribute__((ext_vector_type(16))) _Float16 v16h;
typedef __attribute__((ext_vector_type(8)))  _Float16 v8h;
typedef __attribute__((ext_vector_type(8)))  float    v8f;

#define B_     4
#define T_     4096
#define IN_    1024
#define DIM    512
#define HEADS  8
#define DH     64
#define MLM    256
#define LGRP   17
#define KCONV  33
#define NTOK   4097
#define NPAD   4352
#define PADF   255

#define BK      32
#define LDSTR   40      // halfs per LDS row: 80 bytes -> every b128 access 16B aligned

__device__ inline v8h cvt8(const float4 a, const float4 b, float s)
{
    v8h r;
    r[0] = (_Float16)(a.x * s); r[1] = (_Float16)(a.y * s);
    r[2] = (_Float16)(a.z * s); r[3] = (_Float16)(a.w * s);
    r[4] = (_Float16)(b.x * s); r[5] = (_Float16)(b.y * s);
    r[6] = (_Float16)(b.z * s); r[7] = (_Float16)(b.w * s);
    return r;
}

// ---------------------------------------------------------------------------
// LDS-staged, double-buffered batched WMMA GEMM:
//   C[z] = epi( alphaA * A[z] @ (B[z] or B[z]^T) )
// block = 128 threads (4 waves); block tile 64(M) x 64(N), K-step 32.
// Global f32 tiles are loaded with b128, converted to f16, staged in LDS:
//   As[m_local][k_local]   (A fragments = contiguous 16B runs)
//   Bs[n_local][k_local]   (B fragments = contiguous 16B runs, both transB modes)
// Each wave owns a 16x64 strip -> 4 v_wmma_f32_16x16x32_f16 per K-step with
// fragments rebuilt via 2 ds_load_b128 per operand tile. Double buffering
// overlaps next-tile staging with the current WMMAs (1 barrier per K-step).
// batch index z decomposed as (z>>3, z&7) so (b,h)-strided views into the
// fused QKV tensor work with two strides. All shapes here have M%64==0,
// N%64==0, K%32==0 -> no predication, EXEC all ones as WMMA requires.
// ---------------------------------------------------------------------------
__global__ __launch_bounds__(128)
void gemm_wmma_f16(const float* __restrict__ A, const float* __restrict__ Bm,
                   float* __restrict__ C, const float* __restrict__ bias,
                   int K, int lda, int ldb, int ldc,
                   long long sAb, long long sAh, long long sBb, long long sBh,
                   long long sCb, long long sCh,
                   int transB, float alphaA, int epi,
                   int rpi, int rpo, int roff)
{
    __shared__ _Float16 As[2][64][LDSTR];
    __shared__ _Float16 Bs[2][64][LDSTR];

    const int zbt = blockIdx.z >> 3, zh = blockIdx.z & 7;
    A  += zbt * sAb + zh * sAh;
    Bm += zbt * sBb + zh * sBh;
    C  += zbt * sCb + zh * sCh;

    const int tid  = threadIdx.x;
    const int wave = tid >> 5;
    const int lane = tid & 31;
    const int lm   = lane & 15;
    const int hig  = lane >> 4;          // lane group 0/1
    const int kb   = hig * 8;            // fragment K base
    const int m0   = blockIdx.y * 64;
    const int n0   = blockIdx.x * 64;

    // staging coordinates (128 threads cover a 64x32 f32 tile as 4xfloat4 each)
    const int arow = tid >> 1;           // 0..63 : A row / B column(n) for transB
    const int asel = (tid & 1) * 16;     // k offset 0/16
    const int brow = tid >> 2;           // 0..31 : k row for transB==0
    const int bsel = (tid & 3) * 16;     // n offset for transB==0

    const float* Ag  = A  + (long long)(m0 + arow) * lda + asel;
    const float* Bg1 = Bm + (long long)(n0 + arow) * ldb + asel;   // transB==1
    const float* Bg0 = Bm + (long long)brow * ldb + n0 + bsel;     // transB==0

    auto stage = [&](int k0, int buf) {
        {   // A tile
            const float4 p0 = *(const float4*)(Ag + k0 + 0);
            const float4 p1 = *(const float4*)(Ag + k0 + 4);
            const float4 p2 = *(const float4*)(Ag + k0 + 8);
            const float4 p3 = *(const float4*)(Ag + k0 + 12);
            *(v8h*)&As[buf][arow][asel]     = cvt8(p0, p1, alphaA);
            *(v8h*)&As[buf][arow][asel + 8] = cvt8(p2, p3, alphaA);
        }
        if (transB) {                      // contiguous along k
            const float4 p0 = *(const float4*)(Bg1 + k0 + 0);
            const float4 p1 = *(const float4*)(Bg1 + k0 + 4);
            const float4 p2 = *(const float4*)(Bg1 + k0 + 8);
            const float4 p3 = *(const float4*)(Bg1 + k0 + 12);
            *(v8h*)&Bs[buf][arow][asel]     = cvt8(p0, p1, 1.f);
            *(v8h*)&Bs[buf][arow][asel + 8] = cvt8(p2, p3, 1.f);
        } else {                           // contiguous along n, transpose into Bs[n][k]
            const float* src = Bg0 + (long long)k0 * ldb;
            const float4 q0 = *(const float4*)(src + 0);
            const float4 q1 = *(const float4*)(src + 4);
            const float4 q2 = *(const float4*)(src + 8);
            const float4 q3 = *(const float4*)(src + 12);
            const float t16[16] = { q0.x,q0.y,q0.z,q0.w, q1.x,q1.y,q1.z,q1.w,
                                    q2.x,q2.y,q2.z,q2.w, q3.x,q3.y,q3.z,q3.w };
#pragma unroll
            for (int j = 0; j < 16; ++j)
                Bs[buf][bsel + j][brow] = (_Float16)t16[j];
        }
    };

    v8f acc[4];
#pragma unroll
    for (int t = 0; t < 4; ++t)
#pragma unroll
        for (int e = 0; e < 8; ++e) acc[t][e] = 0.f;

    const int nIter = K / BK;
    stage(0, 0);
    __syncthreads();

    for (int it = 0; it < nIter; ++it) {
        const int buf = it & 1;
        if (it + 1 < nIter) {
            __builtin_prefetch(Ag + (it + 2) * BK, 0, 1);   // L2 prefetch, tile after next
            stage((it + 1) * BK, buf ^ 1);
        }
        const v8h a_lo = *(const v8h*)&As[buf][wave * 16 + lm][kb];
        const v8h a_hi = *(const v8h*)&As[buf][wave * 16 + lm][kb + 16];
        const v16h af = __builtin_shufflevector(a_lo, a_hi,
                            0,1,2,3,4,5,6,7,8,9,10,11,12,13,14,15);
#pragma unroll
        for (int t = 0; t < 4; ++t) {
            const v8h b_lo = *(const v8h*)&Bs[buf][t * 16 + lm][kb];
            const v8h b_hi = *(const v8h*)&Bs[buf][t * 16 + lm][kb + 16];
            const v16h bf = __builtin_shufflevector(b_lo, b_hi,
                                0,1,2,3,4,5,6,7,8,9,10,11,12,13,14,15);
            acc[t] = __builtin_amdgcn_wmma_f32_16x16x32_f16(
                false, af, false, bf, (short)0, acc[t], false, false);
        }
        __syncthreads();
    }

    // Epilogue. C/D layout: vgpr r -> M = m0 + wave*16 + r + hig*8, N = n0+t*16+lm.
    const int mbase = m0 + wave * 16 + hig * 8;
#pragma unroll
    for (int t = 0; t < 4; ++t) {
        const int nc = n0 + t * 16 + lm;
        const float bv = bias ? bias[nc] : 0.f;
#pragma unroll
        for (int r = 0; r < 8; ++r) {
            const int mm = mbase + r;
            float v = acc[t][r] + bv;
            long long crow = mm;
            if (epi == 2) {                 // fc1: relu + posenc + row remap
                v = fmaxf(v, 0.f);
                const int bb = mm / rpi, tt = mm % rpi;
                const float div = __expf((float)(nc & ~1) *
                                         (-9.2103403719761836f / 512.f));
                const float ang = (float)tt * div;
                v += (nc & 1) ? __cosf(ang) : __sinf(ang);
                crow = (long long)bb * rpo + tt + roff;
            }
            C[crow * (long long)ldc + nc] = v;
        }
    }
}

// ---------------------------------------------------------------------------
// LayerNorm over DIM=512 with optional front-zero-padding of the output.
// ---------------------------------------------------------------------------
__global__ __launch_bounds__(256)
void layernorm_pad(const float* __restrict__ in, float* __restrict__ out,
                   const float* __restrict__ g, const float* __restrict__ b,
                   int rows_in, int rows_out, int pad)
{
    const int row  = blockIdx.x;
    const int bidx = row / rows_out;
    const int r    = row % rows_out;
    float* op = out + (long long)row * DIM;
    const int tid = threadIdx.x;
    if (r < pad) { op[tid] = 0.f; op[tid + 256] = 0.f; return; }
    const float* ip = in + ((long long)bidx * rows_in + (r - pad)) * DIM;
    const float x0 = ip[tid], x1 = ip[tid + 256];
    __shared__ float red[256];
    red[tid] = x0 + x1; __syncthreads();
    for (int s = 128; s > 0; s >>= 1) { if (tid < s) red[tid] += red[tid+s]; __syncthreads(); }
    const float mu = red[0] * (1.f / 512.f); __syncthreads();
    const float d0 = x0 - mu, d1 = x1 - mu;
    red[tid] = d0*d0 + d1*d1; __syncthreads();
    for (int s = 128; s > 0; s >>= 1) { if (tid < s) red[tid] += red[tid+s]; __syncthreads(); }
    const float rstd = rsqrtf(red[0] * (1.f / 512.f) + 1e-5f);
    op[tid]       = d0 * rstd * g[tid]       + b[tid];
    op[tid + 256] = d1 * rstd * g[tid + 256] + b[tid + 256];
}

__global__ void set_cls(const float* __restrict__ cls, float* __restrict__ h)
{
    const int idx = blockIdx.x * 256 + threadIdx.x;     // B_*DIM
    const int c = idx & (DIM - 1), b = idx >> 9;
    h[(long long)b * NTOK * DIM + c] = cls[c];
}

// landmark means over groups of LGRP consecutive tokens (scale folded in)
__global__ __launch_bounds__(256)
void landmark_mean(const float* __restrict__ qkv, float* __restrict__ out,
                   int colOff, float scale)
{
    const int idx = blockIdx.x * 256 + threadIdx.x;     // B*H*MLM*DH
    const int d = idx & 63, m = (idx >> 6) & 255, h = (idx >> 14) & 7, b = idx >> 17;
    const float* base = qkv + ((long long)b * NPAD) * (3*DIM) + colOff + h * DH + d;
    float s = 0.f;
#pragma unroll
    for (int j = 0; j < LGRP; ++j) s += base[(long long)(m * LGRP + j) * (3*DIM)];
    out[idx] = s * scale;
}

__global__ __launch_bounds__(256)
void softmax_rows(float* __restrict__ buf, int L)
{
    float* row = buf + (long long)blockIdx.x * L;
    const int tid = threadIdx.x;
    __shared__ float red[256];
    float mx = -1e30f;
    for (int j = tid; j < L; j += 256) mx = fmaxf(mx, row[j]);
    red[tid] = mx; __syncthreads();
    for (int s = 128; s > 0; s >>= 1) { if (tid < s) red[tid] = fmaxf(red[tid], red[tid+s]); __syncthreads(); }
    mx = red[0]; __syncthreads();
    float sum = 0.f;
    for (int j = tid; j < L; j += 256) { const float e = __expf(row[j] - mx); row[j] = e; sum += e; }
    red[tid] = sum; __syncthreads();
    for (int s = 128; s > 0; s >>= 1) { if (tid < s) red[tid] += red[tid+s]; __syncthreads(); }
    const float inv = 1.f / red[0];
    for (int j = tid; j < L; j += 256) row[j] *= inv;
}

// per-(b,h): max row-abs-sum and max col-abs-sum of 256x256 matrix
__global__ __launch_bounds__(256)
void pinv_stats(const float* __restrict__ x, float* __restrict__ stats)
{
    const float* xb = x + (long long)blockIdx.x * MLM * MLM;
    const int tid = threadIdx.x;
    float rs = 0.f, cs = 0.f;
    for (int j = 0; j < MLM; ++j) {
        rs += fabsf(xb[tid * MLM + j]);
        cs += fabsf(xb[j * MLM + tid]);
    }
    __shared__ float r1[256], r2[256];
    r1[tid] = rs; r2[tid] = cs; __syncthreads();
    for (int s = 128; s > 0; s >>= 1) {
        if (tid < s) { r1[tid] = fmaxf(r1[tid], r1[tid+s]); r2[tid] = fmaxf(r2[tid], r2[tid+s]); }
        __syncthreads();
    }
    if (!tid) { stats[blockIdx.x*2] = r1[0]; stats[blockIdx.x*2+1] = r2[0]; }
}

// z0 = x^T / (maxrow*maxcol)
__global__ __launch_bounds__(256)
void pinv_init(const float* __restrict__ x, const float* __restrict__ stats,
               float* __restrict__ z)
{
    const int idx = blockIdx.x * 256 + threadIdx.x;     // B*H*256*256
    const int j = idx & 255, i = (idx >> 8) & 255, zb = idx >> 16;
    const float inv = 1.f / (stats[zb*2] * stats[zb*2+1]);
    z[idx] = x[((long long)zb * MLM + j) * MLM + i] * inv;
}

// out = alpha*I - in   (batched 256x256)
__global__ __launch_bounds__(256)
void alpha_i_minus(const float* __restrict__ in, float* __restrict__ out, float alpha)
{
    const int idx = blockIdx.x * 256 + threadIdx.x;
    const int j = idx & 255, i = (idx >> 8) & 255;
    out[idx] = ((i == j) ? alpha : 0.f) - in[idx];
}

// depthwise K=33 conv of v along the sequence, accumulated into attn output
__global__ __launch_bounds__(256)
void conv_residual(const float* __restrict__ qkv, const float* __restrict__ w,
                   float* __restrict__ aout)
{
    const int idx = blockIdx.x * 256 + threadIdx.x;     // B*H*NPAD*DH
    const int d = idx & 63;
    const int rest = idx >> 6;
    const int n  = rest % NPAD;
    const int zh = rest / NPAD;
    const int h = zh & 7, b = zh >> 3;
    const float* vcol = qkv + ((long long)b * NPAD) * (3*DIM) + 2*DIM + h * DH + d;
    float s = 0.f;
#pragma unroll
    for (int kk = 0; kk < KCONV; ++kk) {
        const int nn = n + kk - KCONV/2;
        if (nn >= 0 && nn < NPAD) s += w[h * KCONV + kk] * vcol[(long long)nn * (3*DIM)];
    }
    aout[idx] += s;
}

// [b,h,n,d] -> [b,n,h*64+d]
__global__ __launch_bounds__(256)
void repack_heads(const float* __restrict__ aout, float* __restrict__ at2)
{
    const int idx = blockIdx.x * 256 + threadIdx.x;     // B*NPAD*DIM
    const int c = idx & (DIM - 1);
    const int rest = idx / DIM;
    const int n = rest % NPAD;
    const int b = rest / NPAD;
    const int h = c >> 6, d = c & 63;
    at2[idx] = aout[(((long long)(b * HEADS + h)) * NPAD + n) * DH + d];
}

// h[b,t,:] += proj[b, PADF+t, :]   (drop front padding)
__global__ __launch_bounds__(256)
void residual_add(float* __restrict__ h, const float* __restrict__ proj)
{
    const int idx = blockIdx.x * 256 + threadIdx.x;     // B*NTOK*DIM
    const int c = idx & (DIM - 1);
    const int rest = idx / DIM;
    const int t = rest % NTOK;
    const int b = rest / NTOK;
    h[idx] += proj[(((long long)b * NPAD) + PADF + t) * DIM + c];
}

__global__ __launch_bounds__(512)
void pool_rows(const float* __restrict__ hn, float* __restrict__ pooled)
{
    const int b = blockIdx.x, c = threadIdx.x;
    const float* base = hn + (long long)b * NTOK * DIM + c;
    float s = 0.f;
    for (int t = 1; t < NTOK; ++t) s += base[(long long)t * DIM];
    pooled[b * DIM + c] = base[0] + s * (1.f / (float)(NTOK - 1));
}

__global__ __launch_bounds__(512)
void head_out(const float* __restrict__ pooled, const float* __restrict__ w,
              const float* __restrict__ bias, float* __restrict__ outp)
{
    const int b = blockIdx.x, c = threadIdx.x;
    __shared__ float red[512];
    red[c] = pooled[b * DIM + c] * w[c]; __syncthreads();
    for (int s = 256; s > 0; s >>= 1) { if (c < s) red[c] += red[c+s]; __syncthreads(); }
    if (!c) {
        const float logit = red[0] + bias[0];
        const float prob  = 1.f / (1.f + __expf(-logit));
        outp[b]     = logit;
        outp[4 + b] = prob;
        outp[8 + b] = (prob > 0.5f) ? 1.f : 0.f;
    }
}

// ---------------------------------------------------------------------------
extern "C" void kernel_launch(void* const* d_in, const int* in_sizes, int n_in,
                              void* d_out, int out_size, void* d_ws, size_t ws_size,
                              hipStream_t stream)
{
    (void)in_sizes; (void)n_in; (void)out_size; (void)ws_size;

    const float* data    = (const float*)d_in[0];
    const float* fc1_w   = (const float*)d_in[1];
    const float* fc1_b   = (const float*)d_in[2];
    const float* cls_tok = (const float*)d_in[3];
    const float* norm_g  = (const float*)d_in[4];
    const float* norm_b  = (const float*)d_in[5];
    const float* fcout_w = (const float*)d_in[6];
    const float* fcout_b = (const float*)d_in[7];
    float* out = (float*)d_out;

    float* ws = (float*)d_ws;
    const long long SZ_H   = (long long)B_ * NTOK * DIM;       // 8,390,656
    const long long SZ_XP  = (long long)B_ * NPAD * DIM;       // 8,912,896
    const long long SZ_QKV = (long long)B_ * NPAD * 3 * DIM;   // 26,738,688
    const long long SZ_L   = (long long)B_ * HEADS * MLM * DH; //   524,288
    const long long SZ_A   = (long long)B_ * HEADS * NPAD * MLM;
    const long long SZ_S   = (long long)B_ * HEADS * MLM * MLM;
    const long long SZ_AO  = (long long)B_ * HEADS * NPAD * DH;

    float* HB   = ws;               ws += SZ_H;
    float* XP   = ws;               ws += SZ_XP;
    float* QKV  = ws;               ws += SZ_QKV;
    float* QL   = ws;               ws += SZ_L;
    float* KL   = ws;               ws += SZ_L;
    float* A1   = ws;               ws += SZ_A;
    float* A3   = ws;               ws += SZ_A;   // reused later as t2
    float* X2   = ws;               ws += SZ_S;
    float* ZB0  = ws;               ws += SZ_S;
    float* ZN0  = ws;               ws += SZ_S;
    float* XZ   = ws;               ws += SZ_S;
    float* TT   = ws;               ws += SZ_S;
    float* T1   = ws;               ws += SZ_L;
    float* AOUT = ws;               ws += SZ_AO;
    float* AT2  = ws;               ws += SZ_XP;
    float* PROJ = ws;               ws += SZ_XP;
    float* STAT = ws;               ws += 64;
    float* POOL = ws;               ws += B_ * DIM;

    auto gemm = [&](const float* A, const float* Bm, float* C, const float* bias,
                    int M, int N, int K, int lda, int ldb, int ldc,
                    long long sAb, long long sAh, long long sBb, long long sBh,
                    long long sCb, long long sCh, int batch,
                    int transB, float alphaA, int epi,
                    int rpi = 1, int rpo = 1, int roff = 0) {
        dim3 g(N / 64, M / 64, batch), blk(128);
        gemm_wmma_f16<<<g, blk, 0, stream>>>(A, Bm, C, bias, K, lda, ldb, ldc,
                                             sAb, sAh, sBb, sBh, sCb, sCh,
                                             transB, alphaA, epi, rpi, rpo, roff);
    };

    const long long SQH = (long long)NPAD * 3 * DIM;   // per-batch qkv stride
    const float SCALE = 0.125f;

    // ---- stem: h = relu(data@fc1_w + b) + posenc, cls token prepended ----
    gemm(data, fc1_w, HB, fc1_b, B_ * T_, DIM, IN_, IN_, DIM, DIM,
         0, 0, 0, 0, 0, 0, 1, 0, 1.f, /*epi=*/2, /*rpi=*/T_, /*rpo=*/NTOK, /*roff=*/1);
    set_cls<<<(B_ * DIM) / 256, 256, 0, stream>>>(cls_tok, HB);

    for (int layer = 0; layer < 2; ++layer) {
        const float* ln_g   = (const float*)d_in[8  + layer * 6];
        const float* ln_b   = (const float*)d_in[9  + layer * 6];
        const float* qkv_w  = (const float*)d_in[10 + layer * 6];
        const float* out_w  = (const float*)d_in[11 + layer * 6];
        const float* out_b  = (const float*)d_in[12 + layer * 6];
        const float* conv_w = (const float*)d_in[13 + layer * 6];

        // xp = zero-pad(front) . layernorm(h)
        layernorm_pad<<<B_ * NPAD, 256, 0, stream>>>(HB, XP, ln_g, ln_b, NTOK, NPAD, PADF);

        // fused QKV projection
        gemm(XP, qkv_w, QKV, nullptr, B_ * NPAD, 3 * DIM, DIM, DIM, 3 * DIM, 3 * DIM,
             0, 0, 0, 0, 0, 0, 1, 0, 1.f, 0);

        // landmarks (q pre-scaled by 1/sqrt(dh))
        landmark_mean<<<SZ_L / 256, 256, 0, stream>>>(QKV, QL, 0,   SCALE / (float)LGRP);
        landmark_mean<<<SZ_L / 256, 256, 0, stream>>>(QKV, KL, DIM, 1.f   / (float)LGRP);

        // a1 = softmax(q*s @ kl^T)   [n x m] per (b,h)
        gemm(QKV, KL, A1, nullptr, NPAD, MLM, DH, 3 * DIM, DH, MLM,
             SQH, DH, 8LL * MLM * DH, (long long)MLM * DH,
             8LL * NPAD * MLM, (long long)NPAD * MLM, B_ * HEADS, 1, SCALE, 0);
        softmax_rows<<<B_ * HEADS * NPAD, 256, 0, stream>>>(A1, MLM);

        // a2 = softmax(ql @ kl^T)    [m x m]
        gemm(QL, KL, X2, nullptr, MLM, MLM, DH, DH, DH, MLM,
             8LL * MLM * DH, (long long)MLM * DH, 8LL * MLM * DH, (long long)MLM * DH,
             8LL * MLM * MLM, (long long)MLM * MLM, B_ * HEADS, 1, 1.f, 0);
        softmax_rows<<<B_ * HEADS * MLM, 256, 0, stream>>>(X2, MLM);

        // a3 = softmax(ql @ k^T)     [m x n]
        gemm(QL, QKV + DIM, A3, nullptr, MLM, NPAD, DH, DH, 3 * DIM, NPAD,
             8LL * MLM * DH, (long long)MLM * DH, SQH, DH,
             8LL * MLM * NPAD, (long long)MLM * NPAD, B_ * HEADS, 1, 1.f, 0);
        softmax_rows<<<B_ * HEADS * MLM, 256, 0, stream>>>(A3, NPAD);

        // iterative Moore-Penrose pinv of a2 (6 Newton iterations, batched)
        pinv_stats<<<B_ * HEADS, 256, 0, stream>>>(X2, STAT);
        pinv_init<<<SZ_S / 256, 256, 0, stream>>>(X2, STAT, ZB0);
        const long long sS = (long long)MLM * MLM;
        float* zc = ZB0; float* zn = ZN0;
        for (int it = 0; it < 6; ++it) {
            gemm(X2, zc, XZ, nullptr, MLM, MLM, MLM, MLM, MLM, MLM,
                 8*sS, sS, 8*sS, sS, 8*sS, sS, B_ * HEADS, 0, 1.f, 0);
            alpha_i_minus<<<SZ_S / 256, 256, 0, stream>>>(XZ, TT, 7.f);
            gemm(XZ, TT, zn, nullptr, MLM, MLM, MLM, MLM, MLM, MLM,
                 8*sS, sS, 8*sS, sS, 8*sS, sS, B_ * HEADS, 0, 1.f, 0);
            alpha_i_minus<<<SZ_S / 256, 256, 0, stream>>>(zn, TT, 15.f);
            gemm(XZ, TT, zn, nullptr, MLM, MLM, MLM, MLM, MLM, MLM,
                 8*sS, sS, 8*sS, sS, 8*sS, sS, B_ * HEADS, 0, 1.f, 0);
            alpha_i_minus<<<SZ_S / 256, 256, 0, stream>>>(zn, TT, 13.f);
            gemm(zc, TT, zn, nullptr, MLM, MLM, MLM, MLM, MLM, MLM,
                 8*sS, sS, 8*sS, sS, 8*sS, sS, B_ * HEADS, 0, 0.25f, 0);
            float* tmp = zc; zc = zn; zn = tmp;
        }

        // t1 = a3 @ v                [m x dh]
        gemm(A3, QKV + 2 * DIM, T1, nullptr, MLM, DH, NPAD, NPAD, 3 * DIM, DH,
             8LL * MLM * NPAD, (long long)MLM * NPAD, SQH, DH,
             8LL * MLM * DH, (long long)MLM * DH, B_ * HEADS, 0, 1.f, 0);
        // t2 = a1 @ pinv(a2)         [n x m]  (reuse A3 buffer)
        gemm(A1, zc, A3, nullptr, NPAD, MLM, MLM, MLM, MLM, MLM,
             8LL * NPAD * MLM, (long long)NPAD * MLM, 8*sS, sS,
             8LL * NPAD * MLM, (long long)NPAD * MLM, B_ * HEADS, 0, 1.f, 0);
        // out = t2 @ t1              [n x dh]
        gemm(A3, T1, AOUT, nullptr, NPAD, DH, MLM, MLM, DH, DH,
             8LL * NPAD * MLM, (long long)NPAD * MLM, 8LL * MLM * DH, (long long)MLM * DH,
             8LL * NPAD * DH, (long long)NPAD * DH, B_ * HEADS, 0, 1.f, 0);

        // depthwise conv residual, head merge, output projection, residual
        conv_residual<<<SZ_AO / 256, 256, 0, stream>>>(QKV, conv_w, AOUT);
        repack_heads<<<SZ_XP / 256, 256, 0, stream>>>(AOUT, AT2);
        gemm(AT2, out_w, PROJ, out_b, B_ * NPAD, DIM, DIM, DIM, DIM, DIM,
             0, 0, 0, 0, 0, 0, 1, 0, 1.f, 0);
        residual_add<<<SZ_H / 256, 256, 0, stream>>>(HB, PROJ);
    }

    // final norm + pooling + classifier head (XP reused for normalized h)
    layernorm_pad<<<B_ * NTOK, 256, 0, stream>>>(HB, XP, norm_g, norm_b, NTOK, NTOK, 0);
    pool_rows<<<B_, 512, 0, stream>>>(XP, POOL);
    head_out<<<B_, 512, 0, stream>>>(POOL, fcout_w, fcout_b, out);
}
